// RadialNCA_72842645340303
// MI455X (gfx1250) — matrix operand order repaired and checked by reading, not compiled, not run on hardware
//
#include <hip/hip_runtime.h>

typedef __attribute__((ext_vector_type(16))) _Float16 v16h;
typedef __attribute__((ext_vector_type(8)))  float    v8f;

#define B_    8
#define H_    256
#define W_    256
#define NPIX  (B_ * H_ * W_)
#define HID   256
#define KPAD  32
#define ALPHA 0.5f
#define TSTEPS 128

// CDNA5 native V_TANH_F32 (TRANS32). Builtin confirmed present for gfx1250.
static __device__ __forceinline__ float fast_tanh(float x) {
#if __has_builtin(__builtin_amdgcn_tanhf)
  return __builtin_amdgcn_tanhf(x);
#else
  float r;
  asm volatile("v_tanh_f32 %0, %1\n\tv_nop" : "=v"(r) : "v"(x));
  return r;
#endif
}

// Convert W1 [27,256] f32 -> w1h [256][32] f16, K padded with zeros.
__global__ void w1_convert_kernel(const float* __restrict__ W1,
                                  _Float16* __restrict__ w1h) {
  int idx = blockIdx.x * blockDim.x + threadIdx.x;
  if (idx >= HID * KPAD) return;
  int n = idx >> 5;
  int k = idx & 31;
  float v = (k < 27) ? W1[k * HID + n] : 0.0f;
  w1h[idx] = (_Float16)v;
}

// Extract channel 0 into a dense plane (ping buffer).
__global__ void init_plane_kernel(const float* __restrict__ state,
                                  float* __restrict__ plane) {
  int p = blockIdx.x * blockDim.x + threadIdx.x;
  if (p < NPIX) plane[p] = state[p * 3];
}

// Reassemble [B,H,W,3]: evolved channel 0 + static channels 1,2.
__global__ void finalize_kernel(const float* __restrict__ state,
                                const float* __restrict__ plane,
                                float* __restrict__ out) {
  int p = blockIdx.x * blockDim.x + threadIdx.x;
  if (p < NPIX) {
    out[p * 3 + 0] = plane[p];
    out[p * 3 + 1] = state[p * 3 + 1];
    out[p * 3 + 2] = state[p * 3 + 2];
  }
}

// One NCA step. Each wave32 handles 16 consecutive pixels (one M tile).
// W1/b1/W2 are staged into LDS once per block (16KB + 2KB out of 320KB/WGP);
// the hot loop feeds v_wmma_f32_16x16x32_f16 from ds_load instead of global,
// removing the exposed global-load latency before each WMMA and cutting L2
// read amplification 8x (per-block staging instead of per-wave streaming).
__global__ void __launch_bounds__(256) nca_step_kernel(
    const float* __restrict__ state,    // original [B,H,W,3] (static ch 1,2)
    const float* __restrict__ pin,      // channel-0 plane, current
    float* __restrict__ pout,           // channel-0 plane, next
    const _Float16* __restrict__ w1h,   // [256][32] f16
    const float* __restrict__ b1,       // [256]
    const float* __restrict__ w2,       // [256]
    const float* __restrict__ b2) {     // [1]
  __shared__ __align__(32) _Float16 sW1[HID * KPAD];  // 16 KB
  __shared__ float sB1[HID];                           // 1 KB
  __shared__ float sW2[HID];                           // 1 KB

  // Cooperative staging: 1024 x 16B chunks of W1, then the two vectors.
  {
    const float4* src = (const float4*)w1h;
    float4* dst = (float4*)sW1;
#pragma unroll
    for (int i = threadIdx.x; i < (HID * KPAD * 2) / 16; i += 256)
      dst[i] = src[i];
    if (threadIdx.x < HID) {            // HID == 256 == blockDim.x
      sB1[threadIdx.x] = b1[threadIdx.x];
      sW2[threadIdx.x] = w2[threadIdx.x];
    }
  }
  __syncthreads();

  const int lane = threadIdx.x & 31;
  const int wave = threadIdx.x >> 5;
  const int tile = blockIdx.x * (blockDim.x >> 5) + wave;
  const int pixelBase = tile * 16;

  const int half = lane >> 4;          // K-half selector (A layout)
  const int m    = lane & 15;          // row M within tile (for A gather)

  const int p   = pixelBase + m;
  const int hw  = H_ * W_;
  const int bb  = p / hw;
  const int rem = p - bb * hw;
  const int y   = rem >> 8;            // W_ == 256
  const int x   = rem & 255;

  // ---- Build A fragment (contiguous K order, same as the B fragment):
  // lane = M (mod 16) + 16*half; element i holds K = i + 16*half.
  // Branchless: clamp coordinates, always load, select 0 when out of bounds.
  v16h a;
#pragma unroll
  for (int i = 0; i < 16; ++i) {
    const int K = i + 16 * half;
    float val = 0.0f;
    if (K < 27) {
      const int kh = K / 9;
      const int kw = (K / 3) % 3;
      const int c  = K % 3;
      const int yy = y + kh - 1;
      const int xx = x + kw - 1;
      const bool inb = (yy >= 0) & (yy < H_) & (xx >= 0) & (xx < W_);
      const int yc = min(max(yy, 0), H_ - 1);
      const int xc = min(max(xx, 0), W_ - 1);
      const int q  = (bb * H_ + yc) * W_ + xc;
      const float* src = (c == 0) ? (pin + q) : (state + q * 3 + c);
      const float ld = *src;
      val = inb ? ld : 0.0f;
    }
    a[i] = (_Float16)val;
  }

  float acc[8];
#pragma unroll
  for (int r = 0; r < 8; ++r) acc[r] = 0.0f;

  const v8f zero = {};
#pragma unroll
  for (int n = 0; n < 16; ++n) {
    const int hidx = n * 16 + m;       // this lane's hidden column
    // B fragment from LDS: 16 consecutive K for column hidx;
    // lanes 0-15 K=0..15, lanes 16-31 K=16..31 (32B-aligned).
    const v16h bm = *(const v16h*)(sW1 + hidx * KPAD + 16 * half);
    v8f c8 = __builtin_amdgcn_wmma_f32_16x16x32_f16(
        false, a, false, bm, (short)0, zero, false, false);
    const float b1v = sB1[hidx];
    const float w2v = sW2[hidx];
#pragma unroll
    for (int r = 0; r < 8; ++r) {
      // D layout: VGPR r, lane l -> element (M = r + 8*(l>>4), N = l&15)
      acc[r] += fast_tanh(c8[r] + b1v) * w2v;
    }
  }

  // Reduce over the 16 N lanes (xor masks < 16 stay within each 16-group).
#pragma unroll
  for (int r = 0; r < 8; ++r) {
#pragma unroll
    for (int mask = 1; mask < 16; mask <<= 1)
      acc[r] += __shfl_xor(acc[r], mask, 32);
  }

  // Lane 0 of each half writes 8 pixels (M = r + 8*half).
  if (m == 0) {
    const float b2v = b2[0];
#pragma unroll
    for (int r = 0; r < 8; ++r) {
      const int pm = pixelBase + r + 8 * half;
      const float oldv = pin[pm];
      float nv = oldv + ALPHA * (acc[r] + b2v);
      nv = fminf(fmaxf(nv, 0.0f), 1.0f);
      pout[pm] = nv;
    }
  }
}

extern "C" void kernel_launch(void* const* d_in, const int* in_sizes, int n_in,
                              void* d_out, int out_size, void* d_ws, size_t ws_size,
                              hipStream_t stream) {
  (void)in_sizes; (void)n_in; (void)out_size; (void)ws_size;

  const float* state = (const float*)d_in[0];   // [8,256,256,3]
  const float* W1    = (const float*)d_in[1];   // [27,256]
  const float* b1    = (const float*)d_in[2];   // [256]
  const float* W2    = (const float*)d_in[3];   // [256,1]
  const float* b2    = (const float*)d_in[4];   // [1]
  float* out = (float*)d_out;

  // Workspace layout: ping plane | pong plane | f16 W1 (all well aligned).
  float* ping = (float*)d_ws;
  float* pong = ping + NPIX;
  _Float16* w1h = (_Float16*)(pong + NPIX);

  w1_convert_kernel<<<(HID * KPAD + 255) / 256, 256, 0, stream>>>(W1, w1h);
  init_plane_kernel<<<(NPIX + 255) / 256, 256, 0, stream>>>(state, ping);

  const int tiles  = NPIX / 16;        // 32768 waves
  const int blocks = tiles / 8;        // 8 waves (256 threads) per block

  float* cur = ping;
  float* nxt = pong;
  for (int t = 0; t < TSTEPS; ++t) {
    nca_step_kernel<<<blocks, 256, 0, stream>>>(state, cur, nxt, w1h, b1, W2, b2);
    float* tmp = cur; cur = nxt; nxt = tmp;
  }

  finalize_kernel<<<(NPIX + 255) / 256, 256, 0, stream>>>(state, cur, out);
}
